// BatchAdaptiveConv2d_18519898980387
// MI455X (gfx1250) — matrix operand order repaired and checked by reading, not compiled, not run on hardware
//
#include <hip/hip_runtime.h>

typedef __attribute__((ext_vector_type(16))) __bf16 v16bf;
typedef __attribute__((ext_vector_type(8)))  float  v8f;

#define Bn    16
#define CIN   32
#define COUT  32
#define Hn    256
#define Wn    256
#define EMB   256
#define TWO_EMB 512
#define BLK_W 128          // output columns per block
#define LCOLS 130          // input columns incl. halo

// ---------------- Kernel 1: adaptation vectors ----------------
// wadapt[b,c] = cat(cond[b],lpe[b]) . wa_w[c,:] + wa_b[c]   (c < 32)
// badapt[b,c] = cat(cond[b],lpe[b]) . ba_w[c,:] + ba_b[c]
// ws layout: [0..511] wadapt (16x32), [512..1023] badapt (16x32)
__global__ __launch_bounds__(64)
void adapt_kernel(const float* __restrict__ cond,
                  const float* __restrict__ lpe,
                  const float* __restrict__ wa_w,
                  const float* __restrict__ wa_b,
                  const float* __restrict__ ba_w,
                  const float* __restrict__ ba_b,
                  float* __restrict__ ws) {
    const int b = blockIdx.x;
    const int t = threadIdx.x;            // 0..63
    const float* w;
    float s;
    float* dst;
    if (t < 32) {
        w   = wa_w + t * TWO_EMB;
        s   = wa_b[t];
        dst = ws + b * CIN + t;
    } else {
        const int c = t - 32;
        w   = ba_w + c * TWO_EMB;
        s   = ba_b[c];
        dst = ws + Bn * CIN + b * COUT + c;
    }
    const float* c0 = cond + b * EMB;
    const float* c1 = lpe  + b * EMB;
    for (int j = 0; j < EMB; ++j) s += c0[j] * w[j];
    for (int j = 0; j < EMB; ++j) s += c1[j] * w[EMB + j];
    *dst = s;
}

// ---------------- Kernel 2: implicit-GEMM grouped conv ----------------
// Block = (batch b, row h, half-row). M = 128 pixels (8 waves x 16),
// N = 32 couts (2 tiles), K = 9 chunks of 32 (ci), one chunk per (dh,dw).
__global__ __launch_bounds__(256)
void conv_kernel(const float* __restrict__ x,        // [CIN][COUT][3][3] weights below
                 const float* __restrict__ weights,  // [CIN][COUT][9]
                 const float* __restrict__ bias,     // [COUT]
                 const float* __restrict__ ws,       // wadapt / badapt
                 float* __restrict__ out) {
    __shared__ __align__(16) __bf16 lds_x[3 * LCOLS * CIN];   // [(row*LCOLS+col)*CIN + ci]
    __shared__ __align__(16) __bf16 lds_w[9 * COUT * CIN];    // [(t*COUT+co)*CIN + ci]

    const int half  = blockIdx.x;         // 0..1
    const int h     = blockIdx.y;         // 0..255
    const int b     = blockIdx.z;         // 0..15
    const int wbase = half * BLK_W;
    const int tid   = threadIdx.x;

    const float* wad = ws + b * CIN;
    const float* bad = ws + Bn * CIN + b * COUT;

    // Stage per-sample modulated weights as bf16, ci-contiguous.
    for (int idx = tid; idx < 9 * COUT * CIN; idx += 256) {
        const int ci   = idx & (CIN - 1);
        const int rest = idx >> 5;
        const int co   = rest & (COUT - 1);
        const int t    = rest >> 5;
        const float wv = weights[(ci * COUT + co) * 9 + t] * wad[ci];
        lds_w[idx] = (__bf16)wv;
    }

    // Stage input tile (rows h-1..h+1, cols wbase-1..wbase+128) as bf16,
    // ci-contiguous in LDS; decode col fastest for coalesced global reads.
    for (int idx = tid; idx < 3 * LCOLS * CIN; idx += 256) {
        const int col  = idx % LCOLS;
        const int rest = idx / LCOLS;
        const int ci   = rest & (CIN - 1);
        const int row  = rest >> 5;
        const int hh   = h + row - 1;
        const int ww   = wbase + col - 1;
        float v = 0.0f;
        if ((unsigned)hh < (unsigned)Hn && (unsigned)ww < (unsigned)Wn)
            v = x[((b * CIN + ci) * Hn + hh) * Wn + ww];
        lds_x[(row * LCOLS + col) * CIN + ci] = (__bf16)v;
    }
    __syncthreads();

    const int lane  = tid & 31;
    const int wave  = tid >> 5;           // 0..7
    const int mbase = wave * 16;          // pixel-tile base inside block
    const int l16   = lane & 15;
    const int hi    = lane >> 4;          // 0 or 1

    v8f acc0 = {};
    v8f acc1 = {};

#pragma unroll
    for (int t = 0; t < 9; ++t) {
        const int dh = t / 3;
        const int dw = t % 3;

        // A fragment: 16x32 bf16. lane<16: M=lane, K=0..15;
        // lane>=16: M=lane-16, K=16..31. K == ci here.
        const int colA = mbase + l16 + dw;                    // 0..129
        const __bf16* ax = &lds_x[(dh * LCOLS + colA) * CIN + hi * 16];
        union { v16bf v; uint4 q[2]; } A;
        A.q[0] = *(const uint4*)(ax);
        A.q[1] = *(const uint4*)(ax + 8);

        // B fragments: 32x16 bf16. lane<16: N=lane, K=0..15 (2 K per dword);
        // lane>=16: N=lane-16, K=16..31.
        const __bf16* bw0 = &lds_w[(t * COUT + l16) * CIN + hi * 16];
        const __bf16* bw1 = &lds_w[(t * COUT + 16 + l16) * CIN + hi * 16];
        union { v16bf v; uint4 q[2]; } B0, B1;
        B0.q[0] = *(const uint4*)(bw0);
        B0.q[1] = *(const uint4*)(bw0 + 8);
        B1.q[0] = *(const uint4*)(bw1);
        B1.q[1] = *(const uint4*)(bw1 + 8);

        acc0 = __builtin_amdgcn_wmma_f32_16x16x32_bf16(
            false, A.v, false, B0.v, (short)0, acc0, false, false);
        acc1 = __builtin_amdgcn_wmma_f32_16x16x32_bf16(
            false, A.v, false, B1.v, (short)0, acc1, false, false);
    }

    // D layout: lane<16 -> N=lane, VGPR r -> M=r; lane>=16 -> N=lane-16, M=r+8.
    const int co0 = l16;
    const int co1 = 16 + l16;
    const float bb0 = bias[co0] * bad[co0];
    const float bb1 = bias[co1] * bad[co1];

    union { v8f v; float f[8]; float4 q[2]; } R0, R1;
    R0.v = acc0;
    R1.v = acc1;
#pragma unroll
    for (int r = 0; r < 8; ++r) { R0.f[r] += bb0; R1.f[r] += bb1; }

    const int colbase = wbase + mbase + hi * 8;   // multiple of 8 -> 32B aligned
    float* o0 = out + ((b * COUT + co0) * Hn + h) * Wn + colbase;
    float* o1 = out + ((b * COUT + co1) * Hn + h) * Wn + colbase;
    *(float4*)(o0)     = R0.q[0];
    *(float4*)(o0 + 4) = R0.q[1];
    *(float4*)(o1)     = R1.q[0];
    *(float4*)(o1 + 4) = R1.q[1];
}

extern "C" void kernel_launch(void* const* d_in, const int* in_sizes, int n_in,
                              void* d_out, int out_size, void* d_ws, size_t ws_size,
                              hipStream_t stream) {
    const float* x    = (const float*)d_in[0];
    const float* cond = (const float*)d_in[1];
    const float* lpe  = (const float*)d_in[2];
    const float* wts  = (const float*)d_in[3];
    const float* bias = (const float*)d_in[4];
    const float* wa_w = (const float*)d_in[5];
    const float* wa_b = (const float*)d_in[6];
    const float* ba_w = (const float*)d_in[7];
    const float* ba_b = (const float*)d_in[8];
    float* out = (float*)d_out;
    float* ws  = (float*)d_ws;   // 1024 floats used

    adapt_kernel<<<Bn, 64, 0, stream>>>(cond, lpe, wa_w, wa_b, ba_w, ba_b, ws);

    dim3 grid(Wn / BLK_W, Hn, Bn);   // 2 x 256 x 16 = 8192 blocks
    conv_kernel<<<grid, 256, 0, stream>>>(x, wts, bias, ws, out);
}
